// MyNet_39900246180039
// MI455X (gfx1250) — matrix operand order repaired and checked by reading, not compiled, not run on hardware
//
#include <hip/hip_runtime.h>

typedef _Float16 half_t;
typedef __attribute__((ext_vector_type(16))) _Float16 v16h;
typedef __attribute__((ext_vector_type(8)))  _Float16 v8h;
typedef __attribute__((ext_vector_type(8)))  float    v8f;
typedef __attribute__((ext_vector_type(4)))  float    v4f;

#define NN       500
#define WWC      360
#define NPAIR    (NN * WWC)        // 180000
#define ROWS_PW  32                // rows (pairs) per wave
#define NTILE32  (NPAIR / ROWS_PW) // 5625
#define WAVES_PB 5                 // 5625 = 5 * 1125 -> exact coverage, EXEC all-1s

// ---------------------------------------------------------------------------
// Generic VALID conv + bias + relu, NCHW/OIHW, one thread per output element.
// Total conv work ~1.3 GFLOP — VALU is fine here.
// ---------------------------------------------------------------------------
__global__ void conv_relu_kernel(const float* __restrict__ in, const float* __restrict__ wt,
                                 const float* __restrict__ bias, float* __restrict__ out,
                                 int total, int Cin, int Hin, int Win,
                                 int Cout, int Ksz, int S, int Hout, int Wout) {
    int idx = blockIdx.x * blockDim.x + threadIdx.x;
    if (idx >= total) return;
    int wo = idx % Wout;
    int ho = (idx / Wout) % Hout;
    int co = (idx / (Wout * Hout)) % Cout;
    int n  = idx / (Wout * Hout * Cout);
    const float* wp = wt + co * Cin * Ksz * Ksz;
    float s = bias[co];
    int hi0 = ho * S, wi0 = wo * S;
    for (int ci = 0; ci < Cin; ++ci) {
        const float* ip = in + ((n * Cin + ci) * Hin + hi0) * Win + wi0;
        for (int kh = 0; kh < Ksz; ++kh)
            for (int kw = 0; kw < Ksz; ++kw)
                s += ip[kh * Win + kw] * wp[(ci * Ksz + kh) * Ksz + kw];
    }
    out[idx] = s > 0.f ? s : 0.f;
}

// ---------------------------------------------------------------------------
// Afeat[n,h] = fc1_w[h,0:512]·x[n] + fc1_w[h,512:519]·states[n] + fc1_b[h]
// ---------------------------------------------------------------------------
__global__ void afeat_kernel(const float* __restrict__ x,       // [N,512]
                             const float* __restrict__ states,  // [N,7]
                             const float* __restrict__ w1,      // [128,522]
                             const float* __restrict__ b1,
                             float* __restrict__ A) {           // [N,128]
    int idx = blockIdx.x * blockDim.x + threadIdx.x;
    if (idx >= NN * 128) return;
    int n = idx / 128, h = idx % 128;
    const float* wr = w1 + h * 522;
    const float* xr = x + n * 512;
    float s = b1[h];
    for (int f = 0; f < 512; ++f) s += xr[f] * wr[f];
    const float* sr = states + n * 7;
    for (int j = 0; j < 7; ++j) s += sr[j] * wr[512 + j];
    A[idx] = s;
}

// Bcw[w,h] = fc1_w[h,519:522]·costweights[w]
__global__ void bcw_kernel(const float* __restrict__ cw,  // [W,3]
                           const float* __restrict__ w1,  // [128,522]
                           float* __restrict__ B) {       // [W,128]
    int idx = blockIdx.x * blockDim.x + threadIdx.x;
    if (idx >= WWC * 128) return;
    int w = idx / 128, h = idx % 128;
    const float* wr = w1 + h * 522 + 519;
    const float* cr = cw + w * 3;
    B[idx] = cr[0] * wr[0] + cr[1] * wr[1] + cr[2] * wr[2];
}

// ---------------------------------------------------------------------------
// Pack [O,K] f32 weights into CDNA5 WMMA B-matrix (32x16 f16) fragments.
// frag[((t*KB+kb)*32+lane)*16+e] = W[t*16+(lane&15), kb*32+(lane>=16?8:0)+(e&7)+(e>=8?16:0)]
// ---------------------------------------------------------------------------
__global__ void pack_frag_kernel(const float* __restrict__ w, half_t* __restrict__ frag,
                                 int O, int Kdim, int total) {
    int idx = blockIdx.x * blockDim.x + threadIdx.x;
    if (idx >= total) return;
    int e    = idx & 15;
    int lane = (idx >> 4) & 31;
    int rest = idx >> 9;             // = t*KB + kb
    int KB   = Kdim / 32;
    int kb   = rest % KB;
    int t    = rest / KB;
    int col  = t * 16 + (lane & 15);
    int k    = kb * 32 + ((lane >> 4) * 8) + (e & 7) + ((e >> 3) * 16);
    float v  = (col < O) ? w[col * Kdim + k] : 0.f;
    frag[idx] = (half_t)v;
}

// ---------------------------------------------------------------------------
// Helpers for the fused MLP
// ---------------------------------------------------------------------------
__device__ __forceinline__ void build_a_frags_l1(const float* __restrict__ Afeat,
                                                 const float* __restrict__ Bcw,
                                                 int p, int half_id, v16h a[4]) {
    // h1 = relu(Afeat[n] + Bcw[w]) directly into A-frag (16-bit 16x32) layout:
    // lane<16 row=lane: e<8 -> K=kb*32+e, e>=8 -> K=kb*32+16+(e-8); lanes>=16: +8.
    int n = p / WWC;
    int w = p - n * WWC;
    const v4f* arow = (const v4f*)(Afeat + n * 128);
    const v4f* brow = (const v4f*)(Bcw   + w * 128);
#pragma unroll
    for (int kb = 0; kb < 4; ++kb) {
        int q0 = kb * 8 + half_id * 2;   // (kb*32 + half_id*8)/4
        v4f x0 = arow[q0]     + brow[q0];
        v4f x1 = arow[q0 + 1] + brow[q0 + 1];
        v4f y0 = arow[q0 + 4] + brow[q0 + 4];
        v4f y1 = arow[q0 + 5] + brow[q0 + 5];
#pragma unroll
        for (int e = 0; e < 4; ++e) {
            a[kb][e]      = (half_t)fmaxf(x0[e], 0.f);
            a[kb][e + 4]  = (half_t)fmaxf(x1[e], 0.f);
            a[kb][e + 8]  = (half_t)fmaxf(y0[e], 0.f);
            a[kb][e + 12] = (half_t)fmaxf(y1[e], 0.f);
        }
    }
}

__device__ __forceinline__ void load_a_frags_lds(const half_t* __restrict__ row,
                                                 int half_id, int nkb, v16h* a) {
    // row = &staging[row_index * Kdim]; loads nkb A-fragments (Kdim = nkb*32)
#pragma unroll
    for (int kb = 0; kb < 4; ++kb) {
        if (kb >= nkb) break;
        int k0 = kb * 32 + half_id * 8;
        v8h lo = *(const v8h*)(row + k0);
        v8h hi = *(const v8h*)(row + k0 + 16);
#pragma unroll
        for (int e = 0; e < 8; ++e) { a[kb][e] = lo[e]; a[kb][e + 8] = hi[e]; }
    }
}

#define WMMA_F16(A, B, C) \
    __builtin_amdgcn_wmma_f32_16x16x32_f16(false, (A), false, (B), (short)0, (C), false, false)

// ---------------------------------------------------------------------------
// Fused MLP over 180,000 pair rows. One wave = one 32-row tile (two 16-row
// WMMA row-tiles sharing every B-fragment load, with B double-buffering).
//   h1 = relu(Afeat[n] + Bcw[w])                   (fc1 factorized away)
//   h2 = relu(h1 @ fc2_w^T + b2)   8 col-tiles x 4 K-steps x 2 row-tiles
//   h3 = relu(h2 @ fc3_w^T + b3)   4 col-tiles x 4 K-steps x 2 row-tiles
//   out = h3 @ fc4_w^T + b4        1 col-tile  x 2 K-steps x 2 row-tiles
// ---------------------------------------------------------------------------
__global__ __launch_bounds__(32 * WAVES_PB)
void mlp_fused_kernel(const float* __restrict__ Afeat,  // [N,128]
                      const float* __restrict__ Bcw,    // [W,128]
                      const half_t* __restrict__ w2f, const float* __restrict__ b2,
                      const half_t* __restrict__ w3f, const float* __restrict__ b3,
                      const half_t* __restrict__ w4f, const float* __restrict__ b4,
                      float* __restrict__ out) {
    __shared__ alignas(16) half_t lds[WAVES_PB][ROWS_PW * 128];   // 8KB/wave staging

    const int wave    = threadIdx.x >> 5;
    const int lane    = threadIdx.x & 31;
    const int half_id = lane >> 4;       // 0: lanes 0-15, 1: lanes 16-31
    const int mrow    = lane & 15;       // A-frag row / D-matrix column
    const int base    = (blockIdx.x * WAVES_PB + wave) * ROWS_PW;

    // Warm WGP$ with the packed weight fragments (global_prefetch_b8).
    for (int o = threadIdx.x * 64; o < 16384; o += 32 * WAVES_PB * 64)
        __builtin_prefetch(w2f + o, 0, 3);
    for (int o = threadIdx.x * 64; o < 8192; o += 32 * WAVES_PB * 64)
        __builtin_prefetch(w3f + o, 0, 3);
    if (threadIdx.x * 64 < 1024) __builtin_prefetch(w4f + threadIdx.x * 64, 0, 3);

    // ---- layer 1: build A-fragments for both 16-row tiles -----------------
    v16h a0[4], a1[4];
    build_a_frags_l1(Afeat, Bcw, base + mrow,      half_id, a0);
    build_a_frags_l1(Afeat, Bcw, base + 16 + mrow, half_id, a1);

    half_t* stg = &lds[wave][0];

    // ---- layer 2: 128 -> 128 (double-buffered B-fragments) ----------------
    v16h bf = *(const v16h*)(w2f + (size_t)lane * 16);
#pragma unroll
    for (int t = 0; t < 8; ++t) {
        v8f acc0 = {}, acc1 = {};
#pragma unroll
        for (int kb = 0; kb < 4; ++kb) {
            int nxt = t * 4 + kb + 1;
            v16h bfn = bf;
            if (nxt < 32)
                bfn = *(const v16h*)(w2f + ((size_t)nxt * 32 + lane) * 16);
            acc0 = WMMA_F16(a0[kb], bf, acc0);
            acc1 = WMMA_F16(a1[kb], bf, acc1);
            bf = bfn;
        }
        float bias = b2[t * 16 + mrow];          // D column = lane&15
#pragma unroll
        for (int r = 0; r < 8; ++r) {            // D row = r + 8*half_id
            int M = r + half_id * 8;
            stg[M * 128 + t * 16 + mrow]        = (half_t)fmaxf(acc0[r] + bias, 0.f);
            stg[(16 + M) * 128 + t * 16 + mrow] = (half_t)fmaxf(acc1[r] + bias, 0.f);
        }
    }
    asm volatile("s_wait_dscnt 0" ::: "memory");  // same-wave DS in-order cross-lane sync

    load_a_frags_lds(stg + mrow * 128,        half_id, 4, a0);
    load_a_frags_lds(stg + (16 + mrow) * 128, half_id, 4, a1);
    asm volatile("" ::: "memory");

    // ---- layer 3: 128 -> 64 ----------------------------------------------
    bf = *(const v16h*)(w3f + (size_t)lane * 16);
#pragma unroll
    for (int t = 0; t < 4; ++t) {
        v8f acc0 = {}, acc1 = {};
#pragma unroll
        for (int kb = 0; kb < 4; ++kb) {
            int nxt = t * 4 + kb + 1;
            v16h bfn = bf;
            if (nxt < 16)
                bfn = *(const v16h*)(w3f + ((size_t)nxt * 32 + lane) * 16);
            acc0 = WMMA_F16(a0[kb], bf, acc0);
            acc1 = WMMA_F16(a1[kb], bf, acc1);
            bf = bfn;
        }
        float bias = b3[t * 16 + mrow];
#pragma unroll
        for (int r = 0; r < 8; ++r) {
            int M = r + half_id * 8;
            stg[M * 64 + t * 16 + mrow]        = (half_t)fmaxf(acc0[r] + bias, 0.f);
            stg[(16 + M) * 64 + t * 16 + mrow] = (half_t)fmaxf(acc1[r] + bias, 0.f);
        }
    }
    asm volatile("s_wait_dscnt 0" ::: "memory");

    load_a_frags_lds(stg + mrow * 64,        half_id, 2, a0);
    load_a_frags_lds(stg + (16 + mrow) * 64, half_id, 2, a1);
    asm volatile("" ::: "memory");

    // ---- layer 4: 64 -> 3 (one padded 16-col tile) ------------------------
    v8f acc0 = {}, acc1 = {};
#pragma unroll
    for (int kb = 0; kb < 2; ++kb) {
        v16h bw = *(const v16h*)(w4f + ((size_t)kb * 32 + lane) * 16);
        acc0 = WMMA_F16(a0[kb], bw, acc0);
        acc1 = WMMA_F16(a1[kb], bw, acc1);
    }
    if (mrow < 3) {                              // valid output columns 0..2
        float bias = b4[mrow];
#pragma unroll
        for (int r = 0; r < 8; ++r) {
            int M = r + half_id * 8;
            out[(base + M) * 3 + mrow]      = acc0[r] + bias;
            out[(base + 16 + M) * 3 + mrow] = acc1[r] + bias;
        }
    }
}

// ---------------------------------------------------------------------------
extern "C" void kernel_launch(void* const* d_in, const int* in_sizes, int n_in,
                              void* d_out, int out_size, void* d_ws, size_t ws_size,
                              hipStream_t stream) {
    const float* costmaps = (const float*)d_in[0];
    const float* states   = (const float*)d_in[1];
    const float* cw       = (const float*)d_in[2];
    const float* c1w = (const float*)d_in[3],  *c1b = (const float*)d_in[4];
    const float* c2w = (const float*)d_in[5],  *c2b = (const float*)d_in[6];
    const float* c3w = (const float*)d_in[7],  *c3b = (const float*)d_in[8];
    const float* c4w = (const float*)d_in[9],  *c4b = (const float*)d_in[10];
    const float* f1w = (const float*)d_in[11], *f1b = (const float*)d_in[12];
    const float* f2w = (const float*)d_in[13], *f2b = (const float*)d_in[14];
    const float* f3w = (const float*)d_in[15], *f3b = (const float*)d_in[16];
    const float* f4w = (const float*)d_in[17], *f4b = (const float*)d_in[18];
    float* out = (float*)d_out;

    char* ws = (char*)d_ws;
    size_t off = 0;
    auto alloc = [&](size_t bytes) -> void* {
        void* p = ws + off;
        off = (off + bytes + 255) & ~(size_t)255;
        return p;
    };
    const int t1 = NN * 32 * 17 * 17;
    const int t2 = NN * 32 * 8 * 8;
    const int t3 = NN * 32 * 6 * 6;
    const int t4 = NN * 32 * 4 * 4;
    float*  act1  = (float*)alloc((size_t)t1 * 4);
    float*  act2  = (float*)alloc((size_t)t2 * 4);
    float*  act3  = (float*)alloc((size_t)t3 * 4);
    float*  act4  = (float*)alloc((size_t)t4 * 4);
    float*  Afeat = (float*)alloc((size_t)NN * 128 * 4);
    float*  Bcw   = (float*)alloc((size_t)WWC * 128 * 4);
    half_t* w2f   = (half_t*)alloc((size_t)16384 * 2);   // 8 tiles x 4 kb x 32 x 16
    half_t* w3f   = (half_t*)alloc((size_t)8192 * 2);    // 4 tiles x 4 kb x 32 x 16
    half_t* w4f   = (half_t*)alloc((size_t)1024 * 2);    // 1 tile  x 2 kb x 32 x 16

    // CNN feature extractor
    conv_relu_kernel<<<(t1 + 255) / 256, 256, 0, stream>>>(costmaps, c1w, c1b, act1, t1, 1, 37, 37, 32, 5, 2, 17, 17);
    conv_relu_kernel<<<(t2 + 255) / 256, 256, 0, stream>>>(act1, c2w, c2b, act2, t2, 32, 17, 17, 32, 3, 2, 8, 8);
    conv_relu_kernel<<<(t3 + 255) / 256, 256, 0, stream>>>(act2, c3w, c3b, act3, t3, 32, 8, 8, 32, 3, 1, 6, 6);
    conv_relu_kernel<<<(t4 + 255) / 256, 256, 0, stream>>>(act3, c4w, c4b, act4, t4, 32, 6, 6, 32, 3, 1, 4, 4);

    // fc1 factorization: n-only and w-only parts
    afeat_kernel<<<(NN * 128 + 255) / 256, 256, 0, stream>>>(act4, states, f1w, f1b, Afeat);
    bcw_kernel<<<(WWC * 128 + 255) / 256, 256, 0, stream>>>(cw, f1w, Bcw);

    // weight fragment packing (B-matrix WMMA layout, f32->f16)
    pack_frag_kernel<<<(16384 + 255) / 256, 256, 0, stream>>>(f2w, w2f, 128, 128, 16384);
    pack_frag_kernel<<<(8192 + 255) / 256, 256, 0, stream>>>(f3w, w3f, 64, 128, 8192);
    pack_frag_kernel<<<(1024 + 255) / 256, 256, 0, stream>>>(f4w, w4f, 3, 64, 1024);

    // fused WMMA MLP over all 180,000 pairs (exact wave coverage, EXEC all-1s)
    mlp_fused_kernel<<<NTILE32 / WAVES_PB, 32 * WAVES_PB, 0, stream>>>(
        Afeat, Bcw, w2f, f2b, w3f, f3b, w4f, f4b, out);
}
